// BiLSTM_36481452212713
// MI455X (gfx1250) — matrix operand order repaired and checked by reading, not compiled, not run on hardware
//
#include <hip/hip_runtime.h>
#include <hip/hip_bf16.h>
#include <math.h>

// ---------------------------------------------------------------------------
// BiLSTM (2-layer, bidirectional, B=64 S=512 D=256 H=512) + FC(1024->128)
// CDNA5 / gfx1250: wave32, WMMA f32_16x16x32_bf16 for the recurrence GEMMs (FC is scalar f32).
//
// Pipeline:
//   1. convert x and build Wcat = [w_ih | w_hh] in bf16, bias = b_ih+b_hh
//   2. layer0 fwd recurrence (512 steps)  -> out0[:, :, 0:512]   (bf16)
//   3. layer0 bwd recurrence (512 steps)  -> out0[:, :, 512:1024] (bf16)
//   4. layer1 fwd recurrence (512 steps)  -> hf1_last (f32)
//   5. layer1 bwd single step (t = S-1, h=c=0) -> hb1_last (f32)
//   6. FC: out = [hf1_last | hb1_last] @ fc_w^T + fc_b
//
// Recurrence kernel: 16 workgroups x 256 threads. WG w owns h-columns
// [32w, 32w+32) of all 4 gate blocks (N-slice of 128 gate columns). Per step:
// WMMA GEMM [64 x 128 x (DX+512)] -> LDS -> LSTM cell -> h ping-pong in
// global (bf16) -> atomic global barrier.
// ---------------------------------------------------------------------------

typedef __bf16 v16bf __attribute__((ext_vector_type(16)));
typedef __bf16 v8bf  __attribute__((ext_vector_type(8)));
typedef float  v8f   __attribute__((ext_vector_type(8)));

#define BATCH   64
#define HID     512
#define SEQ     512
#define GATES   2048   // 4*HID
#define NWG     16     // workgroups in recurrence kernel
#define WGCOLS  32     // h columns owned per WG  (NWG*WGCOLS == HID)
#define NLOC    128    // gate columns per WG (4 blocks * WGCOLS)

// ---------------- helpers: WMMA fragment loads (ISA §7.12.2 layouts) -------

// A 16x32 bf16: lanes 0-15 -> rows M, chunks K[0..7] & K[16..23];
//               lanes 16-31 -> rows M, chunks K[8..15] & K[24..31].
__device__ __forceinline__ v16bf load_a_frag(const __bf16* rowbase, int lane) {
    const int koff = (lane & 16) ? 8 : 0;
    v8bf c0 = *(const v8bf*)(rowbase + koff);
    v8bf c1 = *(const v8bf*)(rowbase + koff + 16);
    return __builtin_shufflevector(c0, c1, 0, 1, 2, 3, 4, 5, 6, 7,
                                            8, 9, 10, 11, 12, 13, 14, 15);
}

// B 32x16 bf16 (column n per lane): lanes 0-15 -> K[0..7] & K[16..23],
// lanes 16-31 -> K[8..15] & K[24..31]. Wcat row = gate column.
__device__ __forceinline__ v16bf load_b_frag(const __bf16* rowbase, int lane) {
    const int koff = (lane & 16) ? 8 : 0;
    v8bf c0 = *(const v8bf*)(rowbase + koff);
    v8bf c1 = *(const v8bf*)(rowbase + koff + 16);
    return __builtin_shufflevector(c0, c1, 0, 1, 2, 3, 4, 5, 6, 7,
                                            8, 9, 10, 11, 12, 13, 14, 15);
}

__device__ __forceinline__ float sigmoidf_(float x) {
    return 1.0f / (1.0f + __expf(-x));
}

// ---------------- conversion / packing kernels -----------------------------

__global__ void k_f32_to_bf16(const float* __restrict__ src,
                              __bf16* __restrict__ dst, int n) {
    int i = blockIdx.x * blockDim.x + threadIdx.x;
    if (i < n) dst[i] = (__bf16)src[i];
}

// Wcat[n][k] : k < DX -> w_ih[n][k], else w_hh[n][k-DX].  n in [0,2048)
__global__ void k_build_wcat(const float* __restrict__ wih,
                             const float* __restrict__ whh,
                             __bf16* __restrict__ wcat, int DX) {
    const int K = DX + HID;
    int i = blockIdx.x * blockDim.x + threadIdx.x;
    if (i >= GATES * K) return;
    int n = i / K;
    int k = i - n * K;
    float v = (k < DX) ? wih[(size_t)n * DX + k]
                       : whh[(size_t)n * HID + (k - DX)];
    wcat[i] = (__bf16)v;
}

__global__ void k_combine_bias(const float* __restrict__ bih,
                               const float* __restrict__ bhh,
                               float* __restrict__ bias) {
    int i = blockIdx.x * blockDim.x + threadIdx.x;
    if (i < GATES) bias[i] = bih[i] + bhh[i];
}

// ---------------- recurrence kernel ----------------------------------------

__global__ __launch_bounds__(256)
void k_lstm_dir(const __bf16* __restrict__ X,    // [B][Stot][DX] layer input
                int DX,
                const __bf16* __restrict__ Wcat, // [2048][DX+512]
                const float* __restrict__ bias,  // [2048]
                __bf16* __restrict__ out_seq,    // [B][Stot][ostride] or null
                int ostride, int ooff,
                __bf16* __restrict__ hping,      // [2][B][512]
                float* __restrict__ cbuf,        // [B][512]
                float* __restrict__ hlast,       // [B][512] f32 or null
                int nsteps, int Stot, int rev,
                int* __restrict__ bar)           // zeroed before launch
{
    __shared__ float Gs[BATCH][NLOC];   // 32 KB: gate pre-activations

    const int wg   = blockIdx.x;        // 0..NWG-1
    const int tid  = threadIdx.x;
    const int lane = tid & 31;
    const int wave = tid >> 5;          // 0..7
    const int Ktot = DX + HID;

    for (int s = 0; s < nsteps; ++s) {
        const int t    = rev ? (Stot - 1 - s) : s;
        const int Kmax = (s == 0) ? DX : Ktot;   // first step: h = 0
        const __bf16* hcur = hping + (size_t)(s & 1) * BATCH * HID;

        // ---- GEMM: G[64][128] = A[64][K] * B[K][128] ----------------------
        for (int i = 0; i < 4; ++i) {
            const int tile = wave * 4 + i;       // 32 tiles
            const int m0   = (tile & 3) << 4;    // 0,16,32,48
            const int nl0  = (tile >> 2) << 4;   // 0..112

            // B row (gate column) for this lane
            const int nl   = nl0 + (lane & 15);
            const int gcol = ((nl >> 5) << 9) + wg * WGCOLS + (nl & 31);
            const __bf16* brow = Wcat + (size_t)gcol * Ktot;

            // A row (batch) for this lane
            const int bm = m0 + (lane & 15);
            const __bf16* xrow = X + ((size_t)bm * Stot + t) * DX;
            const __bf16* hrow = hcur + (size_t)bm * HID;

            v8f acc = {};
            for (int k0 = 0; k0 < Kmax; k0 += 32) {
                const __bf16* abase =
                    (k0 < DX) ? (xrow + k0) : (hrow + (k0 - DX));
                v16bf af = load_a_frag(abase, lane);
                v16bf bf = load_b_frag(brow + k0, lane);
                acc = __builtin_amdgcn_wmma_f32_16x16x32_bf16(
                          false, af, false, bf, (short)0, acc, false, false);
            }

            // C/D layout: lanes 0-15 rows m0+0..7, lanes 16-31 rows m0+8..15
            const int rb = (lane & 16) ? 8 : 0;
            const int cc = nl0 + (lane & 15);
            #pragma unroll
            for (int r = 0; r < 8; ++r)
                Gs[m0 + rb + r][cc] = acc[r];
        }
        __syncthreads();

        // ---- LSTM cell on owned h-slice ----------------------------------
        __bf16* hnext = hping + (size_t)((s + 1) & 1) * BATCH * HID;
        for (int it = tid; it < BATCH * WGCOLS; it += 256) {
            const int b  = it >> 5;
            const int jl = it & 31;
            const int jg = wg * WGCOLS + jl;     // global h column

            float gi = Gs[b][jl]        + bias[jg];
            float gf = Gs[b][32 + jl]   + bias[512 + jg];
            float gg = Gs[b][64 + jl]   + bias[1024 + jg];
            float go = Gs[b][96 + jl]   + bias[1536 + jg];

            gi = sigmoidf_(gi);
            gf = sigmoidf_(gf);
            gg = tanhf(gg);
            go = sigmoidf_(go);

            float c = (s == 0) ? (gi * gg)
                               : (gf * cbuf[(size_t)b * HID + jg] + gi * gg);
            cbuf[(size_t)b * HID + jg] = c;
            float h = go * tanhf(c);

            hnext[(size_t)b * HID + jg] = (__bf16)h;
            if (out_seq)
                out_seq[((size_t)b * Stot + t) * ostride + ooff + jg] =
                    (__bf16)h;
            if (hlast && s == nsteps - 1)
                hlast[(size_t)b * HID + jg] = h;
        }

        // ---- grid-wide barrier (monotonic count, gen = s+1) --------------
        __syncthreads();
        if (tid == 0) {
            __threadfence();
            const int gen = s + 1;
            int arrived = atomicAdd(&bar[0], 1) + 1;
            if (arrived == gen * NWG) atomicExch(&bar[1], gen);
            while (*(volatile int*)&bar[1] < gen) {
                __builtin_amdgcn_s_sleep(2);
            }
        }
        __syncthreads();
    }
}

// ---------------- final FC --------------------------------------------------

__global__ void k_fc(const float* __restrict__ hf,   // [64][512]
                     const float* __restrict__ hb,   // [64][512]
                     const float* __restrict__ fcw,  // [128][1024]
                     const float* __restrict__ fcb,  // [128]
                     float* __restrict__ out)        // [64][128]
{
    int i = blockIdx.x * blockDim.x + threadIdx.x;
    if (i >= BATCH * 128) return;
    int b = i >> 7;
    int o = i & 127;
    const float* w = fcw + (size_t)o * 1024;
    float acc = fcb[o];
    for (int j = 0; j < HID; ++j) {
        acc = fmaf(hf[(size_t)b * HID + j], w[j], acc);
        acc = fmaf(hb[(size_t)b * HID + j], w[512 + j], acc);
    }
    out[i] = acc;
}

// ---------------- host launcher --------------------------------------------

extern "C" void kernel_launch(void* const* d_in, const int* in_sizes, int n_in,
                              void* d_out, int out_size, void* d_ws,
                              size_t ws_size, hipStream_t stream) {
    (void)in_sizes; (void)n_in; (void)out_size; (void)ws_size;

    const float* x    = (const float*)d_in[0];
    const float* wih0f = (const float*)d_in[1];
    const float* whh0f = (const float*)d_in[2];
    const float* bih0f = (const float*)d_in[3];
    const float* bhh0f = (const float*)d_in[4];
    const float* wih0b = (const float*)d_in[5];
    const float* whh0b = (const float*)d_in[6];
    const float* bih0b = (const float*)d_in[7];
    const float* bhh0b = (const float*)d_in[8];
    const float* wih1f = (const float*)d_in[9];
    const float* whh1f = (const float*)d_in[10];
    const float* bih1f = (const float*)d_in[11];
    const float* bhh1f = (const float*)d_in[12];
    const float* wih1b = (const float*)d_in[13];
    const float* whh1b = (const float*)d_in[14];
    const float* bih1b = (const float*)d_in[15];
    const float* bhh1b = (const float*)d_in[16];
    const float* fcw  = (const float*)d_in[17];
    const float* fcb  = (const float*)d_in[18];
    float* out = (float*)d_out;

    // workspace carving (256B aligned), total ~99 MB
    char* ws = (char*)d_ws;
    size_t off = 0;
    auto carve = [&](size_t bytes) -> void* {
        off = (off + 255) & ~(size_t)255;
        void* p = ws + off;
        off += bytes;
        return p;
    };
    const size_t NX   = (size_t)BATCH * SEQ * 256;   // x elems
    __bf16* Xbf    = (__bf16*)carve(NX * 2);
    __bf16* out0   = (__bf16*)carve((size_t)BATCH * SEQ * 1024 * 2);
    __bf16* Wc0f   = (__bf16*)carve((size_t)GATES * 768 * 2);
    __bf16* Wc0b   = (__bf16*)carve((size_t)GATES * 768 * 2);
    __bf16* Wc1f   = (__bf16*)carve((size_t)GATES * 1536 * 2);
    __bf16* Wc1b   = (__bf16*)carve((size_t)GATES * 1536 * 2);
    float*  bs0f   = (float*)carve(GATES * 4);
    float*  bs0b   = (float*)carve(GATES * 4);
    float*  bs1f   = (float*)carve(GATES * 4);
    float*  bs1b   = (float*)carve(GATES * 4);
    __bf16* hping  = (__bf16*)carve((size_t)2 * BATCH * HID * 2);
    float*  cbuf   = (float*)carve((size_t)BATCH * HID * 4);
    float*  hf1    = (float*)carve((size_t)BATCH * HID * 4);
    float*  hb1    = (float*)carve((size_t)BATCH * HID * 4);
    int*    bar    = (int*)carve(256);

    // 1) conversions / packing
    k_f32_to_bf16<<<(int)((NX + 255) / 256), 256, 0, stream>>>(x, Xbf, (int)NX);
    {
        int n0 = GATES * 768, n1 = GATES * 1536;
        k_build_wcat<<<(n0 + 255) / 256, 256, 0, stream>>>(wih0f, whh0f, Wc0f, 256);
        k_build_wcat<<<(n0 + 255) / 256, 256, 0, stream>>>(wih0b, whh0b, Wc0b, 256);
        k_build_wcat<<<(n1 + 255) / 256, 256, 0, stream>>>(wih1f, whh1f, Wc1f, 1024);
        k_build_wcat<<<(n1 + 255) / 256, 256, 0, stream>>>(wih1b, whh1b, Wc1b, 1024);
        k_combine_bias<<<(GATES + 255) / 256, 256, 0, stream>>>(bih0f, bhh0f, bs0f);
        k_combine_bias<<<(GATES + 255) / 256, 256, 0, stream>>>(bih0b, bhh0b, bs0b);
        k_combine_bias<<<(GATES + 255) / 256, 256, 0, stream>>>(bih1f, bhh1f, bs1f);
        k_combine_bias<<<(GATES + 255) / 256, 256, 0, stream>>>(bih1b, bhh1b, bs1b);
    }

    // 2) layer0 forward -> out0[:,:,0:512]
    hipMemsetAsync(bar, 0, 256, stream);
    k_lstm_dir<<<NWG, 256, 0, stream>>>(Xbf, 256, Wc0f, bs0f,
                                        out0, 1024, 0,
                                        hping, cbuf, nullptr,
                                        SEQ, SEQ, 0, bar);
    // 3) layer0 backward -> out0[:,:,512:1024]
    hipMemsetAsync(bar, 0, 256, stream);
    k_lstm_dir<<<NWG, 256, 0, stream>>>(Xbf, 256, Wc0b, bs0b,
                                        out0, 1024, 512,
                                        hping, cbuf, nullptr,
                                        SEQ, SEQ, 1, bar);
    // 4) layer1 forward -> hf1 (final h only)
    hipMemsetAsync(bar, 0, 256, stream);
    k_lstm_dir<<<NWG, 256, 0, stream>>>(out0, 1024, Wc1f, bs1f,
                                        nullptr, 0, 0,
                                        hping, cbuf, hf1,
                                        SEQ, SEQ, 0, bar);
    // 5) layer1 backward: exactly one step at t = S-1 (h=c=0) -> hb1
    hipMemsetAsync(bar, 0, 256, stream);
    k_lstm_dir<<<NWG, 256, 0, stream>>>(out0, 1024, Wc1b, bs1b,
                                        nullptr, 0, 0,
                                        hping, cbuf, hb1,
                                        1, SEQ, 1, bar);

    // 6) FC
    k_fc<<<(BATCH * 128 + 255) / 256, 256, 0, stream>>>(hf1, hb1, fcw, fcb, out);
}
